// DSConv_54717883351526
// MI455X (gfx1250) — matrix operand order repaired and checked by reading, not compiled, not run on hardware
//
#include <hip/hip_runtime.h>

// ---------------------------------------------------------------------------
// Problem constants: f(8,64,128,128) f32; K=9 tap deformable separable conv.
// ---------------------------------------------------------------------------
#define B_   8
#define C_   64
#define W_   128
#define H_   128
#define K_   9
#define OUT_ 64
#define KK_  576          // GEMM K = K_*C_
#define KP_  288          // packed (bf16x2) K
#define KPP  292          // padded LDS row stride (dwords) for A tile
#define NSP  16384        // W_*H_

typedef __attribute__((ext_vector_type(16))) __bf16 bf16x16;
typedef __attribute__((ext_vector_type(8)))  float  f32x8;
typedef __attribute__((ext_vector_type(4)))  unsigned int v4u;
typedef __attribute__((ext_vector_type(8)))  int v8i;
typedef __attribute__((ext_vector_type(4)))  int v4i;

union Frag16 { bf16x16 v; unsigned int u[8]; uint4 q[2]; };

__device__ __forceinline__ unsigned short f2bf(float x) {
  union { float f; unsigned int u; } v; v.f = x;
  unsigned int u = v.u;
  unsigned int r = (u + 0x7fffu + ((u >> 16) & 1u)) >> 16;   // RNE
  return (unsigned short)r;
}

// ---------------------------------------------------------------------------
// K0: zero the stat accumulators (harness poisons ws; must re-zero per call)
// ---------------------------------------------------------------------------
__global__ void k_zero(float* p) {
  int t = threadIdx.x;
  if (t < 320) p[t] = 0.f;   // bn_sums(32) + bn_stats(32) + gn_sums(256)
}

// ---------------------------------------------------------------------------
// K0b: pre-pack w_dsc (OUT,C,K) f32 -> bf16x2 B-panel in per-lane WMMA
//      fragment order: W3[((ks*4+nt)*32 + lane)*8 + v]
//      lane: n = nt*16 + (lane&15), kp = ks*16 + 8*(lane>>4) + v
//      kk = k*64 + c  =>  k = kp>>5, c0 = (kp&31)*2
// ---------------------------------------------------------------------------
__global__ void k_pack_w(const float* __restrict__ w_dsc, unsigned int* __restrict__ w2) {
  int i = blockIdx.x * 256 + threadIdx.x;
  if (i >= KP_ * OUT_) return;
  int v    = i & 7;
  int lane = (i >> 3) & 31;
  int nt   = (i >> 8) & 3;
  int ks   = i >> 10;
  int kp = ks * 16 + 8 * (lane >> 4) + v;
  int n  = nt * 16 + (lane & 15);
  int k  = kp >> 5;
  int c0 = (kp & 31) * 2;
  unsigned short lo = f2bf(w_dsc[(n * C_ + c0)     * K_ + k]);
  unsigned short hi = f2bf(w_dsc[(n * C_ + c0 + 1) * K_ + k]);
  w2[i] = (unsigned int)lo | ((unsigned int)hi << 16);
}

// ---------------------------------------------------------------------------
// K1: 3x3 offset conv (first 9 of 18 channels only) + BN sum/sumsq
//     LDS-tiled 16x16 pixel tile with halo (dynamic LDS: 103,680 B)
// ---------------------------------------------------------------------------
__global__ void k_off_conv(const float* __restrict__ f, const float* __restrict__ w_off,
                           const float* __restrict__ b_off, float* __restrict__ off9,
                           float* __restrict__ bn_sums) {
  extern __shared__ char smem[];
  float* ftile = (float*)smem;          // 64 * 18 * 18
  float* wtile = ftile + 64 * 324;      // 9 * 64 * 9
  __shared__ float bnacc[18];

  const int t   = threadIdx.x;
  const int bid = blockIdx.x;
  const int bx = bid & 7, by = (bid >> 3) & 7, b = bid >> 6;

  for (int i = t; i < 64 * 324; i += 256) {
    int c = i / 324, r = i - c * 324, yy = r / 18, xx = r - yy * 18;
    int gy = by * 16 + yy - 1, gx = bx * 16 + xx - 1;
    float v = 0.f;
    if (gy >= 0 && gy < W_ && gx >= 0 && gx < H_)
      v = f[((b * C_ + c) * W_ + gy) * H_ + gx];
    ftile[i] = v;
  }
  for (int i = t; i < 9 * C_ * 9; i += 256) wtile[i] = w_off[i];  // ch<9 prefix
  if (t < 18) bnacc[t] = 0.f;
  __syncthreads();

  const int yy = t >> 4, xx = t & 15;
  float acc[9];
#pragma unroll
  for (int ch = 0; ch < 9; ++ch) acc[ch] = b_off[ch];

  for (int c = 0; c < C_; ++c) {
    float r0[9];
#pragma unroll
    for (int dy = 0; dy < 3; ++dy)
#pragma unroll
      for (int dx = 0; dx < 3; ++dx)
        r0[dy * 3 + dx] = ftile[c * 324 + (yy + dy) * 18 + (xx + dx)];
#pragma unroll
    for (int ch = 0; ch < 9; ++ch) {
      const float* wp = &wtile[(ch * C_ + c) * 9];
      float s = acc[ch];
#pragma unroll
      for (int tap = 0; tap < 9; ++tap) s = fmaf(r0[tap], wp[tap], s);
      acc[ch] = s;
    }
  }
  const int gy = by * 16 + yy, gx = bx * 16 + xx;
#pragma unroll
  for (int ch = 0; ch < 9; ++ch) {
    off9[((b * K_ + ch) * W_ + gy) * H_ + gx] = acc[ch];
    atomicAdd(&bnacc[ch], acc[ch]);
    atomicAdd(&bnacc[9 + ch], acc[ch] * acc[ch]);
  }
  __syncthreads();
  if (t < 18) atomicAdd(&bn_sums[t], bnacc[t]);
}

// ---------------------------------------------------------------------------
// K2: BN finalize (per-channel mu, rstd)
// ---------------------------------------------------------------------------
__global__ void k_bn_fin(const float* __restrict__ bn_sums, float* __restrict__ bn_stats) {
  int t = threadIdx.x;
  if (t < 9) {
    const float N = (float)(B_ * NSP);
    float mu  = bn_sums[t] / N;
    float var = bn_sums[9 + t] / N - mu * mu;
    bn_stats[t]     = mu;
    bn_stats[9 + t] = rsqrtf(var + 1e-5f);
  }
}

// ---------------------------------------------------------------------------
// K3: BN-affine + tanh + cumulative offsets (reference recurrence, CENTER=4)
// ---------------------------------------------------------------------------
__global__ void k_offsets(const float* __restrict__ off9, const float* __restrict__ bn_stats,
                          const float* __restrict__ gmm, const float* __restrict__ bta,
                          float* __restrict__ offn) {
  int idx = blockIdx.x * 256 + threadIdx.x;   // b*16384 + w*128 + h
  int b = idx >> 14, sp = idx & (NSP - 1);
  float tv[9];
#pragma unroll
  for (int k = 0; k < 9; ++k) {
    float v = off9[(b * K_ + k) * NSP + sp];
    v = (v - bn_stats[k]) * bn_stats[9 + k] * gmm[k] + bta[k];
    tv[k] = tanhf(v);
  }
  float o[9];
  o[4] = 0.f;
  o[5] = tv[5]; o[6] = o[5] + tv[6]; o[7] = o[6] + tv[7];
  o[3] = tv[3]; o[2] = o[3] + tv[2]; o[1] = o[2] + tv[1];
  o[0] = tv[0]; o[8] = tv[8];
#pragma unroll
  for (int k = 0; k < 9; ++k) offn[(b * K_ + k) * NSP + sp] = o[k];
}

// ---------------------------------------------------------------------------
// K4: fused deform-gather -> LDS im2col (bf16) -> WMMA GEMM (M=16,N=64,K=576)
//     block = 128 threads (4 waves), covers (b, w, 64 h's).
//     B panel staged via Tensor Data Mover (tensor_load_to_lds, wave 0);
//     fragments loaded as ds_load_b128 pairs; 18x4 v_wmma fully unrolled.
//     LDS = 157,696 B dynamic.
// ---------------------------------------------------------------------------
__global__ void k_deform_gemm(const float* __restrict__ f, const float* __restrict__ offn,
                              const unsigned int* __restrict__ w2g,
                              const float* __restrict__ b_dsc,
                              float* __restrict__ xpre, float* __restrict__ gn_sums) {
  extern __shared__ char smem[];
  unsigned int* W2s = (unsigned int*)smem;                       // 288*64 dwords (frag order)
  unsigned int* A2s = (unsigned int*)(smem + 73728);             // 64*KPP dwords
  float* wt0 = (float*)(smem + 73728 + 64 * KPP * 4);            // 576
  float* wt1 = wt0 + 576;
  int*   yt0 = (int*)(wt1 + 576);
  int*   yt1 = yt0 + 576;
  __shared__ float gnacc[32];

  const int t   = threadIdx.x;
  const int bid = blockIdx.x;
  const int hb = (bid & 1) * 64;
  const int w  = (bid >> 1) & (W_ - 1);
  const int b  = bid >> 8;

  __builtin_prefetch(w2g, 0, 1);                 // global_prefetch_b8: warm B panel in L2

  // ---- stage bf16x2 weight panel into LDS with the Tensor Data Mover ----
  if ((t >> 5) == 0) {                           // wave 0 only (uniform branch)
    unsigned long long ga = (unsigned long long)(uintptr_t)w2g;  // tile start
    unsigned int ldsoff = __builtin_amdgcn_groupstaticsize();    // dynamic-LDS base = W2s
    v4u g0;
    g0[0] = 1u;                                              // count=1, user descriptor
    g0[1] = ldsoff;                                          // lds_addr (bytes)
    g0[2] = (unsigned int)(ga & 0xffffffffull);              // global_addr[31:0]
    g0[3] = (unsigned int)((ga >> 32) & 0x1ffffffull) | (2u << 30);  // addr[56:32] | type=2
    const unsigned int d0 = KP_ * OUT_;                      // 18432 dwords, 1 row
    v8i g1;
    g1[0] = (int)(2u << 16);                                 // data_size = 4 bytes
    g1[1] = (int)((d0 & 0xffffu) << 16);                     // tensor_dim0[15:0]
    g1[2] = (int)((d0 >> 16) | (1u << 16));                  // tensor_dim0[31:16] | dim1=1
    g1[3] = (int)((d0 & 0xffffu) << 16);                     // dim1 hi=0 | tile_dim0
    g1[4] = (int)(1u);                                       // tile_dim1=1, tile_dim2=0
    g1[5] = (int)d0;                                         // tensor_dim0_stride[31:0]
    g1[6] = 0;                                               // stride0 hi | stride1 lo
    g1[7] = 0;                                               // stride1 hi
    v4i gz4 = {0, 0, 0, 0};
    v8i gz8 = {0, 0, 0, 0, 0, 0, 0, 0};
    __builtin_amdgcn_tensor_load_to_lds(g0, g1, gz4, gz4, gz8, 0);
    __builtin_amdgcn_s_wait_tensorcnt(0);
  }
  if (t < 32) gnacc[t] = 0.f;

  // coordinate/weight tables per (k, h_local).  x = h + k - 4 is integral, so
  // the reference 4-tap bilinear reduces exactly to 2-tap linear along W,
  // with clamped float-index weights reproducing all boundary zeroing.
  for (int i = t; i < K_ * 64; i += 128) {
    int k = i >> 6, hl = i & 63, h = hb + hl;
    float y  = (float)w + offn[(b * K_ + k) * NSP + w * H_ + h];   // EXTEND=1
    int   y0 = (int)floorf(y);
    int y0c = min(max(y0, 0), W_ - 1);
    int y1c = min(max(y0 + 1, 0), W_ - 1);
    float w0 = (float)y1c - y;
    float w1 = y - (float)y0c;
    int x = h + k - 4;
    if (x < 0 || x > H_ - 1) { w0 = 0.f; w1 = 0.f; }
    yt0[i] = y0c; yt1[i] = y1c; wt0[i] = w0; wt1[i] = w1;
  }
  __syncthreads();

  // gather + build bf16x2 A tiles, kk = k*64 + c  (kp = k*32 + c/2)
  const int fb = b * C_ * NSP;
  for (int i = t; i < K_ * 64 * 32; i += 128) {
    int c2 = i & 31, hl = (i >> 5) & 63, k = i >> 11;
    int e  = (k << 6) + hl;
    int xc = min(max(hb + hl + k - 4, 0), H_ - 1);
    float w0 = wt0[e], w1 = wt1[e];
    int base = fb + (c2 * 2) * NSP;
    int o0 = yt0[e] * H_ + xc, o1 = yt1[e] * H_ + xc;
    float s0 = f[base + o0]        * w0 + f[base + o1]        * w1;
    float s1 = f[base + NSP + o0]  * w0 + f[base + NSP + o1]  * w1;
    A2s[hl * KPP + (k * 32 + c2)] = (unsigned int)f2bf(s0) | ((unsigned int)f2bf(s1) << 16);
  }
  __syncthreads();

  // ---- WMMA GEMM (uniform control flow: EXEC all-ones) ----
  const int lane = t & 31, wave = t >> 5;
  const int m = lane & 15, half = lane >> 4;
  const unsigned int* Aw = A2s + wave * 16 * KPP;
  const uint4* Bq = (const uint4*)W2s;           // fragment-order panel

  f32x8 acc[4] = {f32x8{}, f32x8{}, f32x8{}, f32x8{}};
#pragma unroll
  for (int ks = 0; ks < 18; ++ks) {
    Frag16 a;   // 16-bit A 16x32 layout: two contiguous quads per lane
    {
      const uint4* Arow = (const uint4*)&Aw[m * KPP + ks * 16 + 8 * half];
      a.q[0] = Arow[0];        // kp + {0..3}
      a.q[1] = Arow[1];        // kp + 4 + {0..3}
    }
#pragma unroll
    for (int nt = 0; nt < 4; ++nt) {
      Frag16 bf;  // B fragment: 2 x b128 from fragment-order panel
      int fi = ((ks * 4 + nt) * 32 + lane) * 2;
      bf.q[0] = Bq[fi];
      bf.q[1] = Bq[fi + 1];
      acc[nt] = __builtin_amdgcn_wmma_f32_16x16x32_bf16(
          false, a.v, false, bf.v, (short)0, acc[nt], false, false);
    }
  }

  // epilogue: +bias, store x_pre[b][o][w][h], GroupNorm partial sums
  const int hwave = hb + wave * 16 + 8 * half;   // C/D row m = v + 8*half
#pragma unroll
  for (int nt = 0; nt < 4; ++nt) {
    int o = nt * 16 + m;
    float bb = b_dsc[o];
    float s1 = 0.f, s2 = 0.f;
    int obase = ((b * OUT_ + o) * W_ + w) * H_;
#pragma unroll
    for (int v = 0; v < 8; ++v) {
      float x = acc[nt][v] + bb;
      xpre[obase + hwave + v] = x;
      s1 += x; s2 += x * x;
    }
    int g = o >> 2;
    atomicAdd(&gnacc[g * 2],     s1);
    atomicAdd(&gnacc[g * 2 + 1], s2);
  }
  __syncthreads();
  if (t < 32) atomicAdd(&gn_sums[b * 32 + t], gnacc[t]);
}

// ---------------------------------------------------------------------------
// K5: GroupNorm finalize + apply (+ ReLU), float4-vectorized
// ---------------------------------------------------------------------------
__global__ void k_gn_fin(const float* __restrict__ gn_sums, float* __restrict__ gn_stats) {
  int t = threadIdx.x;                       // t = b*16 + g, 128 entries
  if (t < 128) {
    int b = t >> 4, g = t & 15;
    const float N = (float)(4 * NSP);        // 4 channels per group
    float mu  = gn_sums[b * 32 + g * 2] / N;
    float var = gn_sums[b * 32 + g * 2 + 1] / N - mu * mu;
    gn_stats[t]       = mu;
    gn_stats[128 + t] = rsqrtf(var + 1e-5f);
  }
}

__global__ void k_gn_apply(const float* __restrict__ xpre, const float* __restrict__ gn_stats,
                           const float* __restrict__ gmm, const float* __restrict__ bta,
                           float* __restrict__ out) {
  int q = blockIdx.x * 256 + threadIdx.x;    // float4 index over 2M quads
  int b = q >> 18, o = (q >> 12) & 63, g = o >> 2;
  float mu = gn_stats[b * 16 + g], rs = gn_stats[128 + b * 16 + g];
  float ga = gmm[o], be = bta[o];
  float4 x = ((const float4*)xpre)[q];
  x.x = fmaxf((x.x - mu) * rs * ga + be, 0.f);
  x.y = fmaxf((x.y - mu) * rs * ga + be, 0.f);
  x.z = fmaxf((x.z - mu) * rs * ga + be, 0.f);
  x.w = fmaxf((x.w - mu) * rs * ga + be, 0.f);
  ((float4*)out)[q] = x;
}

// ---------------------------------------------------------------------------
// Host launcher — all launches on `stream`, graph-capture safe.
// Workspace layout (floats): off9 | offn | xpre | w2(bf16x2) | stats
// ---------------------------------------------------------------------------
extern "C" void kernel_launch(void* const* d_in, const int* in_sizes, int n_in,
                              void* d_out, int out_size, void* d_ws, size_t ws_size,
                              hipStream_t stream) {
  const float* f     = (const float*)d_in[0];
  const float* w_off = (const float*)d_in[1];
  const float* b_off = (const float*)d_in[2];
  const float* bn_g  = (const float*)d_in[3];
  const float* bn_b  = (const float*)d_in[4];
  const float* w_dsc = (const float*)d_in[5];
  const float* b_dsc = (const float*)d_in[6];
  const float* gn_g  = (const float*)d_in[7];
  const float* gn_b  = (const float*)d_in[8];
  float* out = (float*)d_out;

  float* W        = (float*)d_ws;
  float* off9     = W;                                  // 1,179,648
  float* offn     = W + 1179648;                        // 1,179,648
  float* xpre     = W + 2359296;                        // 8,388,608
  unsigned int* w2 = (unsigned int*)(W + 10747904);     // 18,432 dwords
  float* bn_sums  = W + 10766336;                       // 32
  float* bn_stats = W + 10766368;                       // 32
  float* gn_sums  = W + 10766400;                       // 256
  float* gn_stats = W + 10766656;                       // 256

  k_zero<<<1, 320, 0, stream>>>(bn_sums);
  k_pack_w<<<(KP_ * OUT_ + 255) / 256, 256, 0, stream>>>(w_dsc, w2);
  k_off_conv<<<512, 256, 103680, stream>>>(f, w_off, b_off, off9, bn_sums);
  k_bn_fin<<<1, 32, 0, stream>>>(bn_sums, bn_stats);
  k_offsets<<<512, 256, 0, stream>>>(off9, bn_stats, bn_g, bn_b, offn);
  k_deform_gemm<<<2048, 128, 157696, stream>>>(f, offn, w2, b_dsc, xpre, gn_sums);
  k_gn_fin<<<1, 128, 0, stream>>>(gn_sums, gn_stats);
  k_gn_apply<<<8192, 256, 0, stream>>>(xpre, gn_stats, gn_g, gn_b, out);
}
